// SelfAttention_4655744549078
// MI455X (gfx1250) — matrix operand order repaired and checked by reading, not compiled, not run on hardware
//
#include <hip/hip_runtime.h>

typedef __attribute__((ext_vector_type(16))) _Float16 v16h;
typedef __attribute__((ext_vector_type(8)))  float    v8f;
typedef __attribute__((ext_vector_type(4)))  unsigned v4u;
typedef __attribute__((ext_vector_type(8)))  int      v8i;
typedef __attribute__((ext_vector_type(4)))  int      v4i;
typedef unsigned int u32;

#define L2E 1.4426950408889634f
// B=2 S=2048 D=1024 H=16 E=64

union HF16 { u32 u[8]; v16h h; };
union H8  { unsigned long long q[2]; _Float16 h[8]; };
union H4  { unsigned long long q;    _Float16 h[4]; };

// ---- gfx1250 async global->LDS copy (16B per lane), tracked by ASYNCcnt ----
__device__ __forceinline__ void async_ld_b128(const void* g, void* lds) {
  unsigned loff = (unsigned)(unsigned long long)lds;  // low 32b of flat LDS addr = LDS offset
  asm volatile("global_load_async_to_lds_b128 %0, %1, off"
               :: "v"(loff), "v"(g) : "memory");
}
__device__ __forceinline__ void wait_async1() {
  asm volatile("s_wait_asynccnt 0x1" ::: "memory");
}
__device__ __forceinline__ void wait_async3() {
  asm volatile("s_wait_asynccnt 0x3" ::: "memory");
}
__device__ __forceinline__ void wait_async4() {
  asm volatile("s_wait_asynccnt 0x4" ::: "memory");
}

// ---- Tensor Data Mover: 2D tile (f16) -> LDS, padded rows (16 DW data + 4 DW pad) ----
// D# packing per CDNA5 ISA. Requires tile0 rows of 64B and
// LDS row stride 80B (40 halfwords). dims < 65536.
__device__ __forceinline__ void tdm_load_2d_f16(unsigned lds_off, const void* g,
                                                unsigned tdim0, unsigned tdim1,
                                                unsigned tile0, unsigned tile1,
                                                unsigned stride0) {
  unsigned long long ga = (unsigned long long)g;
  v4u g0 = { 1u,                       // count=1 (valid), user mode
             lds_off,                  // lds_addr
             (u32)ga,                  // global_addr[31:0]
             ((u32)(ga >> 32) & 0x1FFFFFFu) | (2u << 30) };  // addr[56:32] | type=2
  v8i g1 = { (int)((1u << 16)          // data_size = 2B
                 | (1u << 20)          // pad_enable
                 | (3u << 22)          // pad_interval: 16 DWORDs
                 | (3u << 25)),        // pad_amount: 4 DWORDs (16B)
             (int)(tdim0 << 16),       // tensor_dim0[15:0] -> bits[63:48]
             (int)((tdim1 & 0xFFFFu) << 16),  // tensor_dim1[15:0] -> bits[95:80]
             (int)(tile0 << 16),       // tile_dim0 -> bits[127:112]
             (int)tile1,               // tile_dim1 -> bits[143:128]
             (int)stride0,             // tensor_dim0_stride low32
             0, 0 };
  v4i gz = {0, 0, 0, 0};
#if __clang_major__ >= 23
  v8i gz8 = {0, 0, 0, 0, 0, 0, 0, 0};
  __builtin_amdgcn_tensor_load_to_lds(g0, g1, gz, gz, gz8, 0);
#else
  __builtin_amdgcn_tensor_load_to_lds(g0, g1, gz, gz, 0);
#endif
}

// A-matrix fragment (16x32 f16, MxK). lane<16: row=lane, K in {h*8+e | e<8} U {16+h*8+e}
template <typename P>
__device__ __forceinline__ v16h fragA(P row0, int ldh, int lane) {
  const int r = lane & 15, hh = lane >> 4;
  P row = row0 + r * ldh;
  HF16 f;
#pragma unroll
  for (int j = 0; j < 4; ++j) {
    f.u[j]     = *(const u32*)(row + hh * 8 + 2 * j);
    f.u[4 + j] = *(const u32*)(row + 16 + hh * 8 + 2 * j);
  }
  return f.h;
}

// B-matrix fragment (32x16 f16, KxN), operand staged column-major [N][K] stride ldh.
template <typename P>
__device__ __forceinline__ v16h fragB(P col0, int ldh, int lane) {
  const int c = lane & 15, hh = lane >> 4;
  P col = col0 + c * ldh;
  HF16 f;
#pragma unroll
  for (int j = 0; j < 4; ++j) {
    f.u[j]     = *(const u32*)(col + hh * 8 + 2 * j);
    f.u[4 + j] = *(const u32*)(col + 16 + hh * 8 + 2 * j);
  }
  return f.h;
}

__device__ __forceinline__ v8f wmma16(v16h a, v16h b, v8f c) {
  return __builtin_amdgcn_wmma_f32_16x16x32_f16(false, a, false, b, (short)0, c,
                                                false, false);
}

// ---------------- Prep: one-time fp32 -> f16 conversions / transposes ----------------
__global__ void __launch_bounds__(256)
k_cvt_x(const float* __restrict__ x, _Float16* __restrict__ xh) {
  int idx = (blockIdx.x * 256 + threadIdx.x) * 4;  // 4,194,304 total
  float4 v = *(const float4*)(x + idx);
  H4 p; p.h[0] = (_Float16)v.x; p.h[1] = (_Float16)v.y;
  p.h[2] = (_Float16)v.z; p.h[3] = (_Float16)v.w;
  *(H4*)(xh + idx) = p;
}

// wT[((w*16+h)*64+e)*1024 + d] = w_src[w][(h*1024+d)*64 + e]
__global__ void __launch_bounds__(256)
k_cvt_w(const float* __restrict__ wq, const float* __restrict__ wk,
        const float* __restrict__ wv, _Float16* __restrict__ wT) {
  int idx = blockIdx.x * 256 + threadIdx.x;  // 3,145,728 total
  int d = idx & 1023, e = (idx >> 10) & 63, h = (idx >> 16) & 15, w = idx >> 20;
  const float* src = (w == 0) ? wq : (w == 1) ? wk : wv;
  wT[idx] = (_Float16)src[(h * 1024 + d) * 64 + e];
}

// aggT[n*1024 + k] = wagg[k*1024 + n]
__global__ void __launch_bounds__(256)
k_cvt_agg(const float* __restrict__ wagg, _Float16* __restrict__ aggT) {
  int idx = blockIdx.x * 256 + threadIdx.x;  // 1,048,576 total
  int k = idx & 1023, n = idx >> 10;
  aggT[idx] = (_Float16)wagg[k * 1024 + n];
}

// ---------------- Kernel 1: QKV projections ----------------
// grid (32 bh, 16 sblk). Double-buffered: X tile via TDM, W tiles via async-LDS.
__global__ void __launch_bounds__(256)
k_proj(const _Float16* __restrict__ xh, const _Float16* __restrict__ wT,
       _Float16* __restrict__ Qb, _Float16* __restrict__ Kb,
       _Float16* __restrict__ Vt) {
  __shared__ _Float16 Xs[2][128][40];
  __shared__ _Float16 Ws[2][3][64][40];
  const int tid = threadIdx.x, lane = tid & 31, wave = tid >> 5;
  const int bh = blockIdx.x, b = bh >> 4, h = bh & 15;
  const int sBase = blockIdx.y * 128;
  v8f acc[12];  // [ww*4 + nt]
#pragma unroll
  for (int i = 0; i < 12; ++i) acc[i] = (v8f){0.f,0.f,0.f,0.f,0.f,0.f,0.f,0.f};

  const _Float16* xtile0 = xh + (b * 2048 + sBase) * 1024;
  const int we = tid >> 2, wch = tid & 3;

  // issue tile kc into buffer buf
  auto issue = [&](int kc, int buf) {
    if (wave == 0)
      tdm_load_2d_f16((unsigned)(unsigned long long)&Xs[buf][0][0], xtile0 + kc,
                      /*tdim0*/1024, /*tdim1*/4096, /*tile0*/32, /*tile1*/128,
                      /*stride0*/1024);
#pragma unroll
    for (int ww = 0; ww < 3; ++ww)
      async_ld_b128(&wT[((ww * 16 + h) * 64 + we) * 1024 + kc + wch * 8],
                    &Ws[buf][ww][we][wch * 8]);
  };

  issue(0, 0);  // prologue
  for (int kc = 0; kc < 1024; kc += 32) {
    const int cur = (kc >> 5) & 1;
    issue((kc + 32) & 1023, cur ^ 1);   // next tile streams while we compute
    wait_async3();                       // 3 async/wave per tile: current W done
    __builtin_amdgcn_s_wait_tensorcnt(1);  // wave0: current X done; others trivial
    __syncthreads();
    v16h av = fragA(&Xs[cur][16 * wave][0], 40, lane);
    const _Float16* wsbase = &Ws[cur][0][0][0];
    v16h bv = fragB(wsbase, 40, lane);
#pragma unroll
    for (int i = 0; i < 12; ++i) {  // rotate: fetch frag i+1 under WMMA i
      v16h bn = bv;
      if (i + 1 < 12) {
        int ww = (i + 1) >> 2, nt = (i + 1) & 3;
        bn = fragB(wsbase + ww * 2560 + nt * 640, 40, lane);
      }
      acc[i] = wmma16(av, bv, acc[i]);
      bv = bn;
    }
    __syncthreads();
  }
  const int c = lane & 15, hh = lane >> 4;
#pragma unroll
  for (int nt = 0; nt < 4; ++nt)
#pragma unroll
    for (int v = 0; v < 8; ++v) {
      int s = sBase + 16 * wave + v + 8 * hh;
      Qb[(bh * 2048 + s) * 64 + nt * 16 + c] = (_Float16)acc[nt][v];
      Kb[(bh * 2048 + s) * 64 + nt * 16 + c] = (_Float16)acc[4 + nt][v];
    }
  // V written TRANSPOSED [bh][e][t] with packed 16B stores (s runs inside a lane)
#pragma unroll
  for (int nt = 0; nt < 4; ++nt) {
    H8 pk;
#pragma unroll
    for (int v = 0; v < 8; ++v) pk.h[v] = (_Float16)acc[8 + nt][v];
    int e = nt * 16 + c;
    *(H8*)&Vt[(bh * 64 + e) * 2048 + sBase + 16 * wave + 8 * hh] = pk;
  }
}

// ---------------- Kernel 2: per-key (column) softmax stats ----------------
__global__ void __launch_bounds__(256)
k_stats(const _Float16* __restrict__ Qb, const _Float16* __restrict__ Kb,
        float* __restrict__ mg, float* __restrict__ lg) {
  const int tid = threadIdx.x, lane = tid & 31, wave = tid >> 5;
  const int bh = blockIdx.x;
  const int tBase = blockIdx.y * 128 + 16 * wave;
  const _Float16* kbase = Kb + (bh * 2048 + tBase) * 64;
  v16h bk0 = fragB(kbase, 64, lane);
  v16h bk1 = fragB(kbase + 32, 64, lane);
  float m_c = -3.0e38f, l_c = 0.0f;
  for (int sb = 0; sb < 2048; sb += 16) {
    const _Float16* qbase = Qb + (bh * 2048 + sb) * 64;
    v16h a0 = fragA(qbase, 64, lane);
    v16h a1 = fragA(qbase + 32, 64, lane);
    v8f sc = (v8f){0.f,0.f,0.f,0.f,0.f,0.f,0.f,0.f};
    sc = wmma16(a0, bk0, sc);
    sc = wmma16(a1, bk1, sc);
    float bm = -3.0e38f;
#pragma unroll
    for (int i = 0; i < 8; ++i) bm = fmaxf(bm, sc[i] * 0.125f);
    bm = fmaxf(bm, __shfl_xor(bm, 16, 32));
    float ps = 0.0f;
#pragma unroll
    for (int i = 0; i < 8; ++i) ps += exp2f((sc[i] * 0.125f - bm) * L2E);
    ps += __shfl_xor(ps, 16, 32);
    float mn = fmaxf(m_c, bm);
    l_c = l_c * exp2f((m_c - mn) * L2E) + ps * exp2f((bm - mn) * L2E);
    m_c = mn;
  }
  if ((lane >> 4) == 0) {
    int t = bh * 2048 + tBase + (lane & 15);
    mg[t] = m_c;
    lg[t] = l_c;
  }
}

// ---------------- Kernel 3: fold 1/l into V' (in place on transposed V) ----------------
__global__ void __launch_bounds__(256)
k_vscale(_Float16* __restrict__ Vt, const float* __restrict__ lg) {
  int idx = blockIdx.x * 256 + threadIdx.x;  // layout [bh][e][t]
  int t = idx & 2047, bh = idx >> 17;
  float inv = 1.0f / lg[bh * 2048 + t];
  Vt[idx] = (_Float16)((float)Vt[idx] * inv);
}

// ---------------- Kernel 4: flash-style heads = exp(QK^T/8 - m) @ V' ----------------
__global__ void __launch_bounds__(256)
k_attn(const _Float16* __restrict__ Qb, const _Float16* __restrict__ Kb,
       const _Float16* __restrict__ Vt, const float* __restrict__ mg,
       _Float16* __restrict__ multi) {
  __shared__ _Float16 Plds[128][40];
  __shared__ _Float16 Vlds[2][64][40];
  const int tid = threadIdx.x, lane = tid & 31, wave = tid >> 5;
  const int bh = blockIdx.x, b = bh >> 4, h = bh & 15;
  const int sBase = blockIdx.y * 128;
  const int c = lane & 15, hh = lane >> 4;
  const _Float16* qbase = Qb + (bh * 2048 + sBase + 16 * wave) * 64;
  v16h aq0 = fragA(qbase, 64, lane);
  v16h aq1 = fragA(qbase + 32, 64, lane);
  const float* mcolp = mg + bh * 2048;
  v8f acc[4];
#pragma unroll
  for (int i = 0; i < 4; ++i) acc[i] = (v8f){0.f,0.f,0.f,0.f,0.f,0.f,0.f,0.f};
  const int ve = tid >> 2, vch = tid & 3;
  const _Float16* vrow = Vt + (bh * 64 + ve) * 2048 + vch * 8;

  async_ld_b128(vrow, &Vlds[0][ve][vch * 8]);  // prologue: tile tb=0
  for (int tb = 0; tb < 2048; tb += 32) {
    const int cur = (tb >> 5) & 1;
    async_ld_b128(vrow + ((tb + 32) & 2047), &Vlds[cur ^ 1][ve][vch * 8]);
#pragma unroll
    for (int tsub = 0; tsub < 2; ++tsub) {
      const _Float16* kb = Kb + (bh * 2048 + tb + tsub * 16) * 64;
      v16h bk0 = fragB(kb, 64, lane);
      v16h bk1 = fragB(kb + 32, 64, lane);
      v8f sc = (v8f){0.f,0.f,0.f,0.f,0.f,0.f,0.f,0.f};
      sc = wmma16(aq0, bk0, sc);
      sc = wmma16(aq1, bk1, sc);
      float mcol = mcolp[tb + tsub * 16 + c];
#pragma unroll
      for (int v = 0; v < 8; ++v) {
        float p = exp2f((sc[v] * 0.125f - mcol) * L2E);
        Plds[16 * wave + v + 8 * hh][tsub * 16 + c] = (_Float16)p;  // D->A relayout
      }
    }
    wait_async1();  // current V tile resident; next stays in flight
    __syncthreads();
    v16h ap = fragA(&Plds[16 * wave][0], 40, lane);
    const _Float16* vbase = &Vlds[cur][0][0];
    v16h bv = fragB(vbase, 40, lane);
#pragma unroll
    for (int nt = 0; nt < 4; ++nt) {  // rotate fragments under WMMA
      v16h bn = bv;
      if (nt + 1 < 4) bn = fragB(vbase + (nt + 1) * 640, 40, lane);
      acc[nt] = wmma16(ap, bv, acc[nt]);
      bv = bn;
    }
    __syncthreads();
  }
#pragma unroll
  for (int nt = 0; nt < 4; ++nt)
#pragma unroll
    for (int v = 0; v < 8; ++v) {
      int s = sBase + 16 * wave + v + 8 * hh;
      multi[(b * 2048 + s) * 1024 + h * 64 + nt * 16 + c] = (_Float16)acc[nt][v];
    }
}

// ---------------- Kernel 5: out = multi[4096,1024] @ w_agg[1024,1024] ----------------
__device__ __forceinline__ void agg_issue(const _Float16* __restrict__ multi,
                                          const _Float16* __restrict__ aggT,
                                          _Float16 (*A)[40], _Float16 (*Bl)[40],
                                          int mBase, int nBase, int kc, int tid) {
#pragma unroll
  for (int j = 0; j < 2; ++j) {
    int idx = tid + 256 * j;
    int r = idx >> 2, ch = idx & 3;
    async_ld_b128(&multi[(mBase + r) * 1024 + kc + ch * 8], &A[r][ch * 8]);
    async_ld_b128(&aggT[(nBase + r) * 1024 + kc + ch * 8], &Bl[r][ch * 8]);
  }
}

__global__ void __launch_bounds__(256)
k_agg(const _Float16* __restrict__ multi, const _Float16* __restrict__ aggT,
      float* __restrict__ out) {
  __shared__ _Float16 Alds[2][128][40];
  __shared__ _Float16 Blds[2][128][40];
  const int tid = threadIdx.x, lane = tid & 31, wave = tid >> 5;
  const int mBase = blockIdx.x * 128;
  const int nBase = blockIdx.y * 128;
  v8f acc[8];
#pragma unroll
  for (int i = 0; i < 8; ++i) acc[i] = (v8f){0.f,0.f,0.f,0.f,0.f,0.f,0.f,0.f};

  agg_issue(multi, aggT, Alds[0], Blds[0], mBase, nBase, 0, tid);  // prologue
  for (int kc = 0; kc < 1024; kc += 32) {
    const int cur = (kc >> 5) & 1;
    agg_issue(multi, aggT, Alds[cur ^ 1], Blds[cur ^ 1], mBase, nBase,
              (kc + 32) & 1023, tid);
    wait_async4();  // 4 async/wave per tile: current tile done, next in flight
    __syncthreads();
    v16h av = fragA(&Alds[cur][16 * wave][0], 40, lane);
    const _Float16* bbase = &Blds[cur][0][0];
    v16h bv = fragB(bbase, 40, lane);
#pragma unroll
    for (int nt = 0; nt < 8; ++nt) {  // rotate fragments under WMMA
      v16h bn = bv;
      if (nt + 1 < 8) bn = fragB(bbase + (nt + 1) * 640, 40, lane);
      acc[nt] = wmma16(av, bv, acc[nt]);
      bv = bn;
    }
    __syncthreads();
  }
  const int c = lane & 15, hh = lane >> 4;
#pragma unroll
  for (int nt = 0; nt < 8; ++nt)
#pragma unroll
    for (int v = 0; v < 8; ++v)
      out[(mBase + 16 * wave + v + 8 * hh) * 1024 + nBase + nt * 16 + c] = acc[nt][v];
}

extern "C" void kernel_launch(void* const* d_in, const int* in_sizes, int n_in,
                              void* d_out, int out_size, void* d_ws, size_t ws_size,
                              hipStream_t stream) {
  (void)in_sizes; (void)n_in; (void)out_size; (void)ws_size;
  const float* x    = (const float*)d_in[0];
  // d_in[1] = attention_mask (unused, as in reference)
  const float* wq   = (const float*)d_in[2];
  const float* wk   = (const float*)d_in[3];
  const float* wv   = (const float*)d_in[4];
  const float* wagg = (const float*)d_in[5];
  char* ws = (char*)d_ws;
  _Float16* xh   = (_Float16*)(ws);               // 8 MiB  [B,S,D] f16
  _Float16* wT   = (_Float16*)(ws + 8388608);     // 6 MiB  [3,H,E,D] f16
  _Float16* aggT = (_Float16*)(ws + 14680064);    // 2 MiB  [D, H*E] f16 (transposed)
  _Float16* Qb   = (_Float16*)(ws + 16777216);    // 8 MiB  [B,H,S,E]
  _Float16* Kb   = (_Float16*)(ws + 25165824);    // 8 MiB  [B,H,S,E]
  _Float16* Vt   = (_Float16*)(ws + 33554432);    // 8 MiB  [B,H,E,S] (transposed V)
  float*    mg   = (float*)(ws + 41943040);       // 256 KiB [B,H,S]
  float*    lg   = (float*)(ws + 42205184);       // 256 KiB
  _Float16* mu   = (_Float16*)(ws + 42467328);    // 8 MiB  [B,S,H*E]
  float* out = (float*)d_out;

  k_cvt_x  <<<dim3(4096),   256, 0, stream>>>(x, xh);
  k_cvt_w  <<<dim3(12288),  256, 0, stream>>>(wq, wk, wv, wT);
  k_cvt_agg<<<dim3(4096),   256, 0, stream>>>(wagg, aggT);
  k_proj   <<<dim3(32, 16), 256, 0, stream>>>(xh, wT, Qb, Kb, Vt);
  k_stats  <<<dim3(32, 16), 256, 0, stream>>>(Qb, Kb, mg, lg);
  k_vscale <<<dim3(16384),  256, 0, stream>>>(Vt, lg);
  k_attn   <<<dim3(32, 16), 256, 0, stream>>>(Qb, Kb, Vt, mg, mu);
  k_agg    <<<dim3(32, 8),  256, 0, stream>>>(mu, aggT, out);
}
